// PrePrompt_75496935129297
// MI455X (gfx1250) — compile-verified, matrix operand order checked by reading, not run on hardware
//
#include <hip/hip_runtime.h>
#include <hip/hip_bf16.h>
#include <math.h>

// ---------------- problem constants ----------------
constexpr int N_NODES = 307200;
constexpr int N_EDGES = 2457600;
constexpr int HID     = 256;           // N_IN == HIDDEN == 256
constexpr int N_SUB   = 6144;
constexpr int NPS     = 50;            // nodes per subgraph
constexpr int GROUP   = 12;            // SAMPLE_NUM + 2
constexpr int NGROUP  = N_SUB / GROUP; // 512
constexpr int LOGITS  = 2 * HID;       // 512

// ---------------- WMMA types (probe-confirmed) ----------------
typedef __attribute__((ext_vector_type(16))) __bf16 v16bf;
typedef __attribute__((ext_vector_type(8)))  float  v8f;

__device__ inline unsigned short f2bf(float f) {
    unsigned int u = __float_as_uint(f);
    unsigned int r = u + 0x7FFFu + ((u >> 16) & 1u);   // round-to-nearest-even
    return (unsigned short)(r >> 16);
}

// ---------------- pack W (256x256 f32, row-major W[k][n]) into B-fragment layout ----------------
// Per (kt, nt) fragment: 32 lanes x 8 contiguous dwords (32 B per lane -> 2x b128 loads,
// 1 KB contiguous per wave).  Lane `lane` holds column n = nt*16 + (lane&15),
// K-half = (lane>>4)*8 inside the 32-wide K tile; dword p holds bf16 pair (k0, k0+1).
__global__ __launch_bounds__(256) void pack_w_kernel(const float* __restrict__ W1,
                                                     const float* __restrict__ W2,
                                                     unsigned int* __restrict__ Wp) {
    int id = blockIdx.x * blockDim.x + threadIdx.x;        // 0 .. 2*32768-1
    if (id >= 2 * 32768) return;
    int layer = id >> 15;
    int r     = id & 32767;
    int p     = r & 7;
    int lane  = (r >> 3) & 31;
    int nt    = (r >> 8) & 15;
    int kt    = (r >> 12) & 7;
    const float* W = layer ? W2 : W1;
    int n  = nt * 16 + (lane & 15);
    int kh = (lane >> 4) * 8;
    int pk = (p < 4) ? (p * 2 + kh) : (16 + (p - 4) * 2 + kh);
    int k0 = kt * 32 + pk;
    unsigned int lo = f2bf(W[k0 * HID + n]);
    unsigned int hi = f2bf(W[(k0 + 1) * HID + n]);
    Wp[id] = lo | (hi << 16);
}

// ---------------- edge scatter-add: accum[dst] += src_feat[src], one wave per edge ----------------
__global__ __launch_bounds__(256) void scatter_add_kernel(const float* __restrict__ src_feat,
                                                          float* __restrict__ accum,
                                                          const int* __restrict__ ei) {
    int wid  = (blockIdx.x * blockDim.x + threadIdx.x) >> 5;
    int lane = threadIdx.x & 31;
    if (wid >= N_EDGES) return;
    int s = ei[wid];
    int d = ei[N_EDGES + wid];
    const float* sp = src_feat + (size_t)s * HID + lane;
    float*       dp = accum    + (size_t)d * HID + lane;
#pragma unroll
    for (int j = 0; j < 8; ++j)
        atomicAdd(dp + 32 * j, sp[32 * j]);      // coalesced 128B per step
}

// ---------------- fused GEMM + bias + ReLU: Out = relu(Xin @ W + b) ----------------
// Block: 256 threads (8 waves), 64 rows x 256 cols of output.
// A tile (64x256 f32) staged once in LDS as packed bf16 pairs.
// In-place (Out == Xin) is safe: all global reads happen before the barrier.
__global__ __launch_bounds__(256) void gemm_relu_kernel(const float* __restrict__ Xin,
                                                        const unsigned int* __restrict__ Wp,
                                                        const float* __restrict__ bias,
                                                        float* Out) {
    __shared__ unsigned int As[64 * 128];   // 64 rows x 128 bf16-pairs = 32 KB
    const int row0 = blockIdx.x * 64;
    const int tid  = threadIdx.x;

    // stage + convert A tile
    for (int i = tid; i < 64 * 128; i += 256) {
        int r = i >> 7, pr = i & 127;
        float2 v = *(const float2*)(Xin + (size_t)(row0 + r) * HID + pr * 2);
        As[i] = (unsigned int)f2bf(v.x) | ((unsigned int)f2bf(v.y) << 16);
    }
    __syncthreads();

    const int wave = tid >> 5, lane = tid & 31;
    const int wrow = (wave & 3) * 16;       // 0,16,32,48
    const int wcol = (wave >> 2) * 128;     // 0 or 128
    const int lrow = lane & 15;
    const int kh   = lane >> 4;             // 0/1 -> K-half select
    const int q    = kh * 4;                // pair offset within half
    const int nt0  = wcol >> 4;             // first of 8 col tiles for this wave

    v8f acc[8] = {};
#pragma unroll
    for (int kt = 0; kt < 8; ++kt) {
        // A fragment: 2 contiguous 16B groups in LDS -> 2x ds_load_b128
        union { v16bf v; unsigned int u[8]; } A;
        const int abase = (wrow + lrow) * 128 + kt * 16;
#pragma unroll
        for (int p = 0; p < 8; ++p) {
            int pk = (p < 4) ? (p + q) : (8 + (p - 4) + q);
            A.u[p] = As[abase + pk];
        }
        // All 8 B fragments for this k-step: contiguous 32B per lane -> 2x global_load_b128 each
        union { v16bf v; unsigned int u[8]; } B[8];
#pragma unroll
        for (int j = 0; j < 8; ++j) {
            const unsigned int* bp = Wp + ((size_t)(kt * 16 + nt0 + j) * 32 + lane) * 8;
#pragma unroll
            for (int p = 0; p < 8; ++p) B[j].u[p] = bp[p];
        }
#pragma unroll
        for (int j = 0; j < 8; ++j)
            acc[j] = __builtin_amdgcn_wmma_f32_16x16x32_bf16(
                false, A.v, false, B[j].v, (short)0, acc[j], false, false);
    }

    // epilogue: bias + relu, store f32
#pragma unroll
    for (int j = 0; j < 8; ++j) {
        int col = wcol + j * 16 + lrow;
        float bc = bias[col];
#pragma unroll
        for (int v = 0; v < 8; ++v) {
            int row = row0 + wrow + kh * 8 + v;
            float x = acc[j][v] + bc;
            Out[(size_t)row * HID + col] = x > 0.f ? x : 0.f;
        }
    }
}

// ---------------- subgraph mean pooling of concat([h1,h2]) ----------------
__global__ __launch_bounds__(256) void pool_kernel(const float* __restrict__ H1,
                                                   const float* __restrict__ H2,
                                                   float* __restrict__ sums) {
    int idx = blockIdx.x * blockDim.x + threadIdx.x;   // sub*512 + col
    if (idx >= N_SUB * LOGITS) return;
    int sub = idx >> 9, col = idx & 511;
    const float* src = (col < HID) ? (H1 + (size_t)sub * NPS * HID + col)
                                   : (H2 + (size_t)sub * NPS * HID + (col - HID));
    float s = 0.f;
#pragma unroll 5
    for (int r = 0; r < NPS; ++r) s += src[(size_t)r * HID];
    sums[idx] = s * (1.0f / 50.0f);
}

// ---------------- per-group contrastive loss ----------------
__global__ __launch_bounds__(256) void group_loss_kernel(const float* __restrict__ sub,
                                                         float* __restrict__ gloss) {
    const int t = blockIdx.x;                     // 0..511
    const float* base = sub + (size_t)t * GROUP * LOGITS;
    __shared__ float red[256];
    __shared__ float dotv[GROUP];
    __shared__ float nrm2[GROUP];
    const int tid = threadIdx.x;

    for (int k = 0; k < GROUP; ++k) {
        float d = 0.f, n = 0.f;
        for (int i = tid; i < LOGITS; i += 256) {
            float vs = base[i];
            float vk = base[k * LOGITS + i];
            d += vs * vk;
            n += vk * vk;
        }
        red[tid] = d; __syncthreads();
        for (int off = 128; off > 0; off >>= 1) { if (tid < off) red[tid] += red[tid + off]; __syncthreads(); }
        if (tid == 0) dotv[k] = red[0];
        __syncthreads();
        red[tid] = n; __syncthreads();
        for (int off = 128; off > 0; off >>= 1) { if (tid < off) red[tid] += red[tid + off]; __syncthreads(); }
        if (tid == 0) nrm2[k] = red[0];
        __syncthreads();
    }
    if (tid == 0) {
        float ns = sqrtf(nrm2[0]); ns = ns > 1e-8f ? ns : 1e-8f;
        float sim[11];
        for (int j = 0; j < 11; ++j) {
            float nk = sqrtf(nrm2[j + 1]); nk = nk > 1e-8f ? nk : 1e-8f;
            sim[j] = dotv[j + 1] / (ns * nk);
        }
        float den = 0.f;
        for (int j = 1; j < 11; ++j) den += expf(sim[j]);
        gloss[t] = logf(den) - sim[0];
    }
}

__global__ void finalize_kernel(const float* __restrict__ gloss, float* __restrict__ out) {
    __shared__ float s[NGROUP];
    int t = threadIdx.x;
    s[t] = gloss[t];
    __syncthreads();
    for (int off = NGROUP / 2; off > 0; off >>= 1) {
        if (t < off) s[t] += s[t + off];
        __syncthreads();
    }
    if (t == 0) out[0] = s[0] * (1.0f / NGROUP);
}

// ---------------- host launch ----------------
extern "C" void kernel_launch(void* const* d_in, const int* in_sizes, int n_in,
                              void* d_out, int out_size, void* d_ws, size_t ws_size,
                              hipStream_t stream) {
    const float* feature = (const float*)d_in[0];
    const float* W1      = (const float*)d_in[1];
    const float* b1      = (const float*)d_in[2];
    const float* W2      = (const float*)d_in[3];
    const float* b2      = (const float*)d_in[4];
    const int*   ei      = (const int*)d_in[5];
    float* out = (float*)d_out;

    const size_t NF = (size_t)N_NODES * HID;            // 78,643,200 floats
    float*        X     = (float*)d_ws;                 // x = h + agg (then h2 in-place)
    float*        H     = X + NF;                       // h1
    unsigned int* Wp    = (unsigned int*)(H + NF);      // 2 * 32768 dwords
    float*        sums  = (float*)(Wp + 2 * 32768);     // 6144 x 512
    float*        gloss = sums + (size_t)N_SUB * LOGITS;

    const int scatterBlocks = N_EDGES / 8;              // 8 waves of 32 per 256-thread block
    const int gemmBlocks    = N_NODES / 64;

    pack_w_kernel<<<(2 * 32768 + 255) / 256, 256, 0, stream>>>(W1, W2, Wp);

    // Layer 1: X = feature; X[dst] += feature[src]; H = relu(X @ W1 + b1)
    hipMemcpyAsync(X, feature, NF * sizeof(float), hipMemcpyDeviceToDevice, stream);
    scatter_add_kernel<<<scatterBlocks, 256, 0, stream>>>(feature, X, ei);
    gemm_relu_kernel<<<gemmBlocks, 256, 0, stream>>>(X, Wp, b1, H);

    // Layer 2: X = H; X[dst] += H[src]; X = relu(X @ W2 + b2) in-place (h2)
    hipMemcpyAsync(X, H, NF * sizeof(float), hipMemcpyDeviceToDevice, stream);
    scatter_add_kernel<<<scatterBlocks, 256, 0, stream>>>(H, X, ei);
    gemm_relu_kernel<<<gemmBlocks, 256, 0, stream>>>(X, Wp + 32768, b2, X);

    // Pool + loss
    pool_kernel<<<(N_SUB * LOGITS + 255) / 256, 256, 0, stream>>>(H, X, sums);
    group_loss_kernel<<<NGROUP, 256, 0, stream>>>(sums, gloss);
    finalize_kernel<<<1, NGROUP, 0, stream>>>(gloss, out);
}
